// Dir_Encoder_GAT_75814762709158
// MI455X (gfx1250) — matrix operand order repaired and checked by reading, not compiled, not run on hardware
//
#include <hip/hip_runtime.h>
#include <hip/hip_bf16.h>
#include <math.h>

typedef __attribute__((ext_vector_type(16))) _Float16 v16h;
typedef __attribute__((ext_vector_type(8)))  _Float16 v8h;
typedef __attribute__((ext_vector_type(8)))  float    v8f;

#define NN   50000
#define NE   800000
#define NE2  (NE + NN)
#define DIN  128
#define DH   64
#define DOUT 32

__device__ __forceinline__ void atomicMaxF(float* addr, float v) {
    // ordered-int trick: works for mixed signs, init must be -inf
    if (v >= 0.f) atomicMax((int*)addr, __float_as_int(v));
    else          atomicMin((unsigned int*)addr, __float_as_uint(v));
}

// ---------- fp32 -> fp16 convert ----------
__global__ void k_f2h(const float* __restrict__ in, _Float16* __restrict__ out, long long n) {
    long long i = (long long)blockIdx.x * blockDim.x + threadIdx.x;
    if (i < n) out[i] = (_Float16)in[i];
}

// ---------- pack W [N,K] row-major f32 into WMMA B-fragment order ----------
// frag[((kt*ntiles+nt)*32 + lane)*16 + i] = W[nt*16 + (lane&15)][kt*32 + (i>>3)*16 + (lane>>4)*8 + (i&7)]
__global__ void k_pack_w(const float* __restrict__ W, _Float16* __restrict__ wf, int K, int N) {
    int ntiles = N >> 4, ktiles = K >> 5;
    int total  = ktiles * ntiles * 512;
    int idx = blockIdx.x * blockDim.x + threadIdx.x;
    if (idx >= total) return;
    int i    = idx & 15;
    int lane = (idx >> 4) & 31;
    int t    = idx >> 9;
    int nt   = t % ntiles, kt = t / ntiles;
    int col  = nt * 16 + (lane & 15);
    int k    = kt * 32 + (i >> 3) * 16 + (lane >> 4) * 8 + (i & 7);
    wf[idx] = (_Float16)W[(size_t)col * K + k];
}

// ---------- dual GEMM: xl = xh@Wl^T+bl, xr = xh@Wr^T+br via WMMA ----------
__global__ void k_gemm_dual(const _Float16* __restrict__ xh,
                            const _Float16* __restrict__ wlf,
                            const _Float16* __restrict__ wrf,
                            const float* __restrict__ bl,
                            const float* __restrict__ br,
                            float* __restrict__ xl, float* __restrict__ xr,
                            int M, int K, int N) {
    int ntiles = N >> 4;
    int ktiles = K >> 5;
    int wave = (int)((blockIdx.x * (long long)blockDim.x + threadIdx.x) >> 5);
    int lane = threadIdx.x & 31;
    int total = (M >> 4) * ntiles;
    if (wave >= total) return;               // wave-uniform: EXEC all-ones inside
    int mt = wave / ntiles, nt = wave % ntiles;
    int half = lane >> 4;
    int row  = mt * 16 + (lane & 15);        // A: lanes 0-15 and 16-31 both hold M=0..15
    const _Float16* arow = xh + (size_t)row * K + half * 8;

    v8f accl = {}; v8f accr = {};
    for (int kt = 0; kt < ktiles; ++kt) {
        v8h lo = *(const v8h*)(arow + kt * 32);        // K = half*8 + 0..7
        v8h hi = *(const v8h*)(arow + kt * 32 + 16);   // K = half*8 + 16..23
        v16h a;
#pragma unroll
        for (int i = 0; i < 8; ++i) { a[i] = lo[i]; a[i + 8] = hi[i]; }
        const _Float16* bp = wlf + ((size_t)(kt * ntiles + nt) * 32 + lane) * 16;
        const _Float16* bq = wrf + ((size_t)(kt * ntiles + nt) * 32 + lane) * 16;
        v16h bL = *(const v16h*)bp;
        v16h bR = *(const v16h*)bq;
        accl = __builtin_amdgcn_wmma_f32_16x16x32_f16(false, a, false, bL, (short)0, accl, false, false);
        accr = __builtin_amdgcn_wmma_f32_16x16x32_f16(false, a, false, bR, (short)0, accr, false, false);
    }
    int col = nt * 16 + (lane & 15);
    int r0  = mt * 16 + half * 8;            // C: VGPR j -> row j + 8*(lane>=16)
    float bLv = bl[col], bRv = br[col];
#pragma unroll
    for (int j = 0; j < 8; ++j) {
        xl[(size_t)(r0 + j) * N + col] = accl[j] + bLv;
        xr[(size_t)(r0 + j) * N + col] = accr[j] + bRv;
    }
}

// ---------- per-node init ----------
__global__ void k_init4(float* m, float* s, float* cnt, float* asum, int n) {
    int i = blockIdx.x * blockDim.x + threadIdx.x;
    if (i < n) { m[i] = -INFINITY; s[i] = 0.f; cnt[i] = 0.f; asum[i] = 0.f; }
}
__global__ void k_init_ms(float* m, float* s, int n) {
    int i = blockIdx.x * blockDim.x + threadIdx.x;
    if (i < n) { m[i] = -INFINITY; s[i] = 0.f; }
}
__global__ void k_init_out(float* out, const float* __restrict__ bias, int n, int D) {
    long long i = (long long)blockIdx.x * blockDim.x + threadIdx.x;
    if (i < (long long)n * D) out[i] = bias[i % D];
}

// ---------- edge list head + degree/weight-sum counts ----------
__global__ void k_edges_head(const long long* __restrict__ ei, const float* __restrict__ ew,
                             int* __restrict__ src2, int* __restrict__ dst2,
                             float* __restrict__ attr2,
                             float* cnt, float* asum, int E) {
    int e = blockIdx.x * blockDim.x + threadIdx.x;
    if (e >= E) return;
    int s = (int)ei[e];
    int d = (int)ei[(long long)E + e];
    float w = ew[e];
    src2[e] = s; dst2[e] = d; attr2[e] = w;
    atomicAdd(&cnt[d], 1.f);
    atomicAdd(&asum[d], w);
}

// ---------- self-loop tail: mean incoming edge weight ----------
__global__ void k_loop_tail(const float* __restrict__ cnt, const float* __restrict__ asum,
                            int* __restrict__ src2, int* __restrict__ dst2,
                            float* __restrict__ attr2, int E, int n) {
    int i = blockIdx.x * blockDim.x + threadIdx.x;
    if (i >= n) return;
    src2[E + i] = i; dst2[E + i] = i;
    attr2[E + i] = asum[i] / fmaxf(cnt[i], 1.f);
}

// ---------- attention scores: one wave per edge ----------
__global__ void k_score(const float* __restrict__ xl, const float* __restrict__ xr,
                        const float* __restrict__ We, const float* __restrict__ att,
                        const int* __restrict__ src, const int* __restrict__ dst,
                        const float* __restrict__ attr,
                        float* __restrict__ score, float* m, int E2, int D) {
    int wid  = (int)((blockIdx.x * (long long)blockDim.x + threadIdx.x) >> 5);
    int lane = threadIdx.x & 31;
    if (wid >= E2) return;
    int s = src[wid], d = dst[wid];
    float w = attr[wid];
    float p = 0.f;
    for (int c = lane; c < D; c += 32) {
        float v = xl[(size_t)s * D + c] + xr[(size_t)d * D + c] + w * We[c];
        v = v > 0.f ? v : 0.2f * v;               // leaky_relu(0.2)
        p += att[c] * v;
    }
#pragma unroll
    for (int off = 16; off > 0; off >>= 1) p += __shfl_xor(p, off, 32);
    if (lane == 0) { score[wid] = p; atomicMaxF(&m[d], p); }
}

// ---------- exp(score - max) and per-dst sum ----------
__global__ void k_expsum(float* __restrict__ score, const int* __restrict__ dst,
                         const float* __restrict__ m, float* s, int E2) {
    int e = blockIdx.x * blockDim.x + threadIdx.x;
    if (e >= E2) return;
    int d = dst[e];
    float ev = __expf(score[e] - m[d]);
    score[e] = ev;                                // reuse buffer
    atomicAdd(&s[d], ev);
}

// ---------- alpha-weighted scatter: one wave per edge ----------
__global__ void k_scatter(const float* __restrict__ xl,
                          const int* __restrict__ src, const int* __restrict__ dst,
                          const float* __restrict__ expsc, const float* __restrict__ ssum,
                          float* out, int E2, int D) {
    int wid  = (int)((blockIdx.x * (long long)blockDim.x + threadIdx.x) >> 5);
    int lane = threadIdx.x & 31;
    if (wid >= E2) return;
    int s = src[wid], d = dst[wid];
    float alpha = expsc[wid] / ssum[d];
    for (int c = lane; c < D; c += 32)
        atomicAdd(&out[(size_t)d * D + c], alpha * xl[(size_t)s * D + c]);
}

// ---------- ELU in place + f16 copy for next layer ----------
__global__ void k_elu_half(float* __restrict__ h, _Float16* __restrict__ hh, long long n) {
    long long i = (long long)blockIdx.x * blockDim.x + threadIdx.x;
    if (i >= n) return;
    float v = h[i];
    v = v > 0.f ? v : (__expf(v) - 1.f);
    h[i] = v;
    hh[i] = (_Float16)v;
}

// ---------- softplus + eps in place ----------
__global__ void k_softplus(float* __restrict__ z, long long n) {
    long long i = (long long)blockIdx.x * blockDim.x + threadIdx.x;
    if (i >= n) return;
    float v = z[i];
    float r = v > 20.f ? v : (v < -20.f ? __expf(v) : log1pf(__expf(v)));
    z[i] = r + 1e-4f;
}

extern "C" void kernel_launch(void* const* d_in, const int* in_sizes, int n_in,
                              void* d_out, int out_size, void* d_ws, size_t ws_size,
                              hipStream_t stream) {
    (void)in_sizes; (void)n_in; (void)out_size; (void)ws_size;
    const float*     x    = (const float*)d_in[0];
    const long long* ei   = (const long long*)d_in[1];   // int64 per reference
    const float*     ew   = (const float*)d_in[2];
    const float*     Wl1  = (const float*)d_in[3];
    const float*     bl1  = (const float*)d_in[4];
    const float*     Wr1  = (const float*)d_in[5];
    const float*     br1  = (const float*)d_in[6];
    const float*     We1  = (const float*)d_in[7];
    const float*     att1 = (const float*)d_in[8];
    const float*     bias1= (const float*)d_in[9];
    const float*     Wl2  = (const float*)d_in[10];
    const float*     bl2  = (const float*)d_in[11];
    const float*     Wr2  = (const float*)d_in[12];
    const float*     br2  = (const float*)d_in[13];
    const float*     We2  = (const float*)d_in[14];
    const float*     att2 = (const float*)d_in[15];
    const float*     bias2= (const float*)d_in[16];
    float* zout = (float*)d_out;

    char* p = (char*)d_ws;
    auto alloc = [&](size_t b) { void* r = (void*)p; p += (b + 255) & ~(size_t)255; return r; };
    _Float16* xh   = (_Float16*)alloc((size_t)NN * DIN * 2);
    _Float16* hh   = (_Float16*)alloc((size_t)NN * DH * 2);
    float* xl1  = (float*)alloc((size_t)NN * DH * 4);
    float* xr1  = (float*)alloc((size_t)NN * DH * 4);
    float* h    = (float*)alloc((size_t)NN * DH * 4);
    float* xl2  = (float*)alloc((size_t)NN * DOUT * 4);
    float* xr2  = (float*)alloc((size_t)NN * DOUT * 4);
    int*   src2 = (int*)alloc((size_t)NE2 * 4);
    int*   dst2 = (int*)alloc((size_t)NE2 * 4);
    float* attr2= (float*)alloc((size_t)NE2 * 4);
    float* sc   = (float*)alloc((size_t)NE2 * 4);
    float* mbuf = (float*)alloc((size_t)NN * 4);
    float* sbuf = (float*)alloc((size_t)NN * 4);
    float* cnt  = (float*)alloc((size_t)NN * 4);
    float* asum = (float*)alloc((size_t)NN * 4);
    _Float16* wl1f = (_Float16*)alloc((size_t)(DIN/32)*(DH/16)  * 512 * 2);
    _Float16* wr1f = (_Float16*)alloc((size_t)(DIN/32)*(DH/16)  * 512 * 2);
    _Float16* wl2f = (_Float16*)alloc((size_t)(DH/32) *(DOUT/16)* 512 * 2);
    _Float16* wr2f = (_Float16*)alloc((size_t)(DH/32) *(DOUT/16)* 512 * 2);

    const int B = 256;
    auto gb = [](long long n, int b) { return (int)((n + b - 1) / b); };

    // ---- shared prep ----
    k_f2h<<<gb((long long)NN * DIN, B), B, 0, stream>>>(x, xh, (long long)NN * DIN);
    k_pack_w<<<gb((DIN/32)*(DH/16)*512,   B), B, 0, stream>>>(Wl1, wl1f, DIN, DH);
    k_pack_w<<<gb((DIN/32)*(DH/16)*512,   B), B, 0, stream>>>(Wr1, wr1f, DIN, DH);
    k_pack_w<<<gb((DH/32)*(DOUT/16)*512,  B), B, 0, stream>>>(Wl2, wl2f, DH, DOUT);
    k_pack_w<<<gb((DH/32)*(DOUT/16)*512,  B), B, 0, stream>>>(Wr2, wr2f, DH, DOUT);
    k_init4<<<gb(NN, B), B, 0, stream>>>(mbuf, sbuf, cnt, asum, NN);
    k_edges_head<<<gb(NE, B), B, 0, stream>>>(ei, ew, src2, dst2, attr2, cnt, asum, NE);
    k_loop_tail<<<gb(NN, B), B, 0, stream>>>(cnt, asum, src2, dst2, attr2, NE, NN);

    // ---- layer 1 ----
    k_gemm_dual<<<gb((long long)(NN/16)*(DH/16)*32, B), B, 0, stream>>>(
        xh, wl1f, wr1f, bl1, br1, xl1, xr1, NN, DIN, DH);
    k_init_out<<<gb((long long)NN * DH, B), B, 0, stream>>>(h, bias1, NN, DH);
    k_score<<<gb((long long)NE2 * 32, B), B, 0, stream>>>(
        xl1, xr1, We1, att1, src2, dst2, attr2, sc, mbuf, NE2, DH);
    k_expsum<<<gb(NE2, B), B, 0, stream>>>(sc, dst2, mbuf, sbuf, NE2);
    k_scatter<<<gb((long long)NE2 * 32, B), B, 0, stream>>>(
        xl1, src2, dst2, sc, sbuf, h, NE2, DH);
    k_elu_half<<<gb((long long)NN * DH, B), B, 0, stream>>>(h, hh, (long long)NN * DH);

    // ---- layer 2 (accumulate straight into d_out) ----
    k_gemm_dual<<<gb((long long)(NN/16)*(DOUT/16)*32, B), B, 0, stream>>>(
        hh, wl2f, wr2f, bl2, br2, xl2, xr2, NN, DH, DOUT);
    k_init_ms<<<gb(NN, B), B, 0, stream>>>(mbuf, sbuf, NN);
    k_init_out<<<gb((long long)NN * DOUT, B), B, 0, stream>>>(zout, bias2, NN, DOUT);
    k_score<<<gb((long long)NE2 * 32, B), B, 0, stream>>>(
        xl2, xr2, We2, att2, src2, dst2, attr2, sc, mbuf, NE2, DOUT);
    k_expsum<<<gb(NE2, B), B, 0, stream>>>(sc, dst2, mbuf, sbuf, NE2);
    k_scatter<<<gb((long long)NE2 * 32, B), B, 0, stream>>>(
        xl2, src2, dst2, sc, sbuf, zout, NE2, DOUT);
    k_softplus<<<gb((long long)NN * DOUT, B), B, 0, stream>>>(zout, (long long)NN * DOUT);
}
